// MultiHeadAttentionBlock_8598524526937
// MI455X (gfx1250) — compile-verified, matrix operand order checked by reading, not run on hardware
//
#include <hip/hip_runtime.h>
#include <hip/hip_bf16.h>
#include <stdint.h>

// MHA block for MI455X (gfx1250, wave32, WMMA + TDM), round 4.
// Pipeline:
//   0) cvt_bf16:    one-time fp32->bf16 conversion of q,k,v,Wq,Wk,Wv,Wo
//   1) gemm<BF16>:  Qp = q@Wq^T, Kp = k@Wk^T  (bf16 out, 16x64 tile/wave)
//   2) gemm<VT>:    Vt = per-head-transpose(v@Wv^T)  (bf16, [b,h,d,s])
//   3) rope_swap:   Qr = rope(Kp), Kr = rope(Qp)     (reference swaps q/k!)
//   4) attn:        flash attention; K/V tiles DMA'd to LDS via TENSOR_LOAD_TO_LDS
//                   (double-buffered, TENSORcnt-tracked), WMMA from LDS fragments
//   5) gemm<F32>:   out = mh @ Wo^T (fp32 out)

#define BB 4
#define SS 2048
#define DD 768
#define HH 12
#define DKK 64
#define BS (BB * SS)

typedef __attribute__((ext_vector_type(4)))  float        v4f;
typedef __attribute__((ext_vector_type(8)))  float        v8f;
typedef __attribute__((ext_vector_type(8)))  __bf16       v8bf;
typedef __attribute__((ext_vector_type(16))) __bf16       v16bf;
typedef __attribute__((ext_vector_type(4)))  unsigned int u32x4;
typedef __attribute__((ext_vector_type(8)))  int          i32x8;
typedef __attribute__((ext_vector_type(4)))  int          i32x4;

union Frag {
    v16bf v;
    v8bf  h[2];
};

__device__ __forceinline__ v8bf cvt8(v4f a, v4f b) {
    v8bf r;
    r[0] = (__bf16)a[0]; r[1] = (__bf16)a[1]; r[2] = (__bf16)a[2]; r[3] = (__bf16)a[3];
    r[4] = (__bf16)b[0]; r[5] = (__bf16)b[1]; r[6] = (__bf16)b[2]; r[7] = (__bf16)b[3];
    return r;
}

// 16x32 bf16 A/B fragment from a row whose fast axis is the contraction dim.
// ISA layout: elems 0..7 <-> K = half*8 + 0..7 ; elems 8..15 <-> K = 16 + half*8 + 0..7
__device__ __forceinline__ Frag mk_frag_bf16(const __bf16* p, int half) {
    Frag f;
    f.h[0] = *(const v8bf*)(p + half * 8);
    f.h[1] = *(const v8bf*)(p + 16 + half * 8);
    return f;
}

__device__ __forceinline__ float redmax16(float v) {
#pragma unroll
    for (int m = 1; m < 16; m <<= 1) v = fmaxf(v, __shfl_xor(v, m, 32));
    return v;
}
__device__ __forceinline__ float redsum16(float v) {
#pragma unroll
    for (int m = 1; m < 16; m <<= 1) v += __shfl_xor(v, m, 32);
    return v;
}

// ---- Tensor Data Mover: 2D tile (bf16) global -> LDS, per ISA D# (§8.3-8.6) --------
// tile_d0 elems along contiguous dim0 (row stride = stride0 elems), tile_d1 rows.
__device__ __forceinline__ void tdm_load_2d(const __bf16* lds_ptr, const __bf16* gptr,
                                            uint32_t tensor_d0, uint32_t tensor_d1,
                                            uint32_t tile_d0, uint32_t tile_d1,
                                            uint32_t stride0) {
    uint64_t ga = (uint64_t)(uintptr_t)gptr;
    uint32_t lo = (uint32_t)(uintptr_t)lds_ptr;   // flat LDS addr[31:0] == LDS offset
    u32x4 g0;
    g0[0] = 1u;                                            // count=1 (valid user D#)
    g0[1] = lo;                                            // lds_addr (bytes)
    g0[2] = (uint32_t)ga;                                  // global_addr[31:0]
    g0[3] = (uint32_t)((ga >> 32) & 0x01FFFFFFu) | 0x80000000u; // addr[56:32]|type=2
    i32x8 g1;
    g1[0] = (int)(1u << 16);                               // data_size=1 (2 bytes)
    g1[1] = (int)((tensor_d0 & 0xFFFFu) << 16);            // tensor_dim0[15:0]
    g1[2] = (int)((tensor_d0 >> 16) | ((tensor_d1 & 0xFFFFu) << 16));
    g1[3] = (int)((tensor_d1 >> 16) | (tile_d0 << 16));    // tile_dim0
    g1[4] = (int)tile_d1;                                  // tile_dim1 (tile_dim2=0)
    g1[5] = (int)stride0;                                  // tensor_dim0_stride[31:0]
    g1[6] = 0;
    g1[7] = 0;
    i32x4 z4 = {0, 0, 0, 0};                               // groups 2/3: 2D only
    i32x8 z8 = {0, 0, 0, 0, 0, 0, 0, 0};
    __builtin_amdgcn_tensor_load_to_lds(g0, g1, z4, z4, z8, 0);
}

// ---------------- One-time fp32 -> bf16 conversion (8 elems / thread) ----------------
__global__ __launch_bounds__(256) void cvt_bf16(const float* __restrict__ src,
                                                __bf16* __restrict__ dst, int n8) {
    int i = blockIdx.x * blockDim.x + threadIdx.x;
    if (i >= n8) return;
    const v4f* p = (const v4f*)src + (size_t)i * 2;
    ((v8bf*)dst)[i] = cvt8(p[0], p[1]);
}

// ---------------- GEMM: C[m][n] = sum_k X[m][k]*W[n][k], 16x64 tile per wave --------
// MODE 0: bf16 row-major out. MODE 1: fp32 row-major out. MODE 2: Vt[b,h,d,s] bf16.
template <int MODE>
__global__ __launch_bounds__(128) void gemm_tile(const __bf16* __restrict__ X,
                                                 const __bf16* __restrict__ W,
                                                 void* __restrict__ OutP) {
    int wave = threadIdx.x >> 5, lane = threadIdx.x & 31;
    int N = lane & 15, half = lane >> 4;
    int tile = blockIdx.x * 4 + wave;            // (BS/16)*(DD/64) = 6144 tiles
    int m0 = (tile / (DD / 64)) * 16;
    int n0 = (tile % (DD / 64)) * 64;
    const __bf16* arow = X + (size_t)(m0 + N) * DD;
    const __bf16* wrow = W + (size_t)(n0 + N) * DD;
    v8f acc[4];
#pragma unroll
    for (int t = 0; t < 4; t++) acc[t] = (v8f){};
    for (int k0 = 0; k0 < DD; k0 += 32) {
        Frag a = mk_frag_bf16(arow + k0, half);  // reused across 4 WMMAs
#pragma unroll
        for (int t = 0; t < 4; t++) {
            Frag b = mk_frag_bf16(wrow + (size_t)t * 16 * DD + k0, half);
            acc[t] = __builtin_amdgcn_wmma_f32_16x16x32_bf16(false, a.v, false, b.v,
                                                             (short)0, acc[t], false, false);
        }
    }
    if (MODE == 0) {
        __bf16* Out = (__bf16*)OutP;
#pragma unroll
        for (int t = 0; t < 4; t++)
#pragma unroll
            for (int r = 0; r < 8; r++)
                Out[(size_t)(m0 + r + 8 * half) * DD + n0 + t * 16 + N] = (__bf16)acc[t][r];
    } else if (MODE == 1) {
        float* Out = (float*)OutP;
#pragma unroll
        for (int t = 0; t < 4; t++)
#pragma unroll
            for (int r = 0; r < 8; r++)
                Out[(size_t)(m0 + r + 8 * half) * DD + n0 + t * 16 + N] = acc[t][r];
    } else {
        __bf16* Vt = (__bf16*)OutP;
        int h = n0 >> 6;                          // n0 is a multiple of 64
#pragma unroll
        for (int t = 0; t < 4; t++) {
            int d = t * 16 + N;
#pragma unroll
            for (int r = 0; r < 8; r++) {
                int row = m0 + r + 8 * half;      // global (b*S + s)
                int b = row >> 11, s = row & (SS - 1);
                Vt[(((size_t)b * HH + h) * DKK + d) * SS + s] = (__bf16)acc[t][r];
            }
        }
    }
}

// ---------------- RoPE with reference's q/k swap: Qr=rope(Kp), Kr=rope(Qp) ----------
__global__ __launch_bounds__(256) void rope_swap(const __bf16* __restrict__ Qp,
                                                 const __bf16* __restrict__ Kp,
                                                 __bf16* __restrict__ Qr,
                                                 __bf16* __restrict__ Kr) {
    int idx = blockIdx.x * blockDim.x + threadIdx.x;   // B*H*S*32 threads
    int i = idx & 31;
    int t = idx >> 5;
    int s = t & (SS - 1);
    int bh = t >> 11;
    int b = bh / HH, h = bh % HH;
    // theta_i = 10000^(-i/32) = exp(-i * ln(1e4)/32)
    float theta = expf(-(float)i * 0.28782313662425575f);
    float ang = (float)s * theta;
    float cs = cosf(ang), sn = sinf(ang);
    size_t src = ((size_t)b * SS + s) * DD + h * DKK + i;
    float k0v = (float)Kp[src], k1v = (float)Kp[src + 32];
    float q0v = (float)Qp[src], q1v = (float)Qp[src + 32];
    size_t dst = ((size_t)bh * SS + s) * DKK + i;
    Qr[dst]      = (__bf16)(k0v * cs - k1v * sn);   // effective query = rope(K proj)
    Qr[dst + 32] = (__bf16)(k1v * cs + k0v * sn);
    Kr[dst]      = (__bf16)(q0v * cs - q1v * sn);   // effective key   = rope(Q proj)
    Kr[dst + 32] = (__bf16)(q1v * cs + q0v * sn);
}

// ---------------- Flash attention with TDM-staged K/V tiles -------------------------
// Block = 4 waves, 4 consecutive 16-row i-tiles of the SAME (b,h): K/V j-tiles are
// shared, DMA'd once per block into double-buffered LDS by wave 0 (TENSORcnt).
// LDS map (bf16 elems): [0,8192) K bufs (2x 64x64), [8192,16384) V bufs (2x 64x64),
//                       [16384,20480) P staging (4 waves x 16x64).
__global__ __launch_bounds__(128) void attn_kernel(const __bf16* __restrict__ Qr,
                                                   const __bf16* __restrict__ Kr,
                                                   const __bf16* __restrict__ Vt,
                                                   const int* __restrict__ mask,
                                                   __bf16* __restrict__ mh) {
    __shared__ __align__(16) __bf16 smem[2 * 64 * 64 + 2 * 64 * 64 + 4 * 16 * 64];
    int wave = threadIdx.x >> 5, lane = threadIdx.x & 31;
    int N = lane & 15, half = lane >> 4;
    int tile = blockIdx.x * 4 + wave;               // B*H*(S/16) tiles
    int i0 = (tile & (SS / 16 - 1)) * 16;
    int bh = tile >> 7;                             // same for all 4 waves in block
    int b = bh / HH, h = bh % HH;
    const __bf16* Qbase = Qr + (size_t)bh * SS * DKK;
    const __bf16* Kbase = Kr + (size_t)bh * SS * DKK;
    const __bf16* Vbase = Vt + (size_t)bh * DKK * SS;
    const int* mrow = mask + b * SS;
    __bf16* pl = smem + 16384 + wave * 1024;

    // Loop-invariant Q fragments (dk chunks 0 and 32)
    Frag aq0, aq1;
    {
        const __bf16* qrow = Qbase + (size_t)(i0 + N) * DKK;
        aq0 = mk_frag_bf16(qrow, half);
        aq1 = mk_frag_bf16(qrow + 32, half);
    }

    float mrun[8], lrun[8];
    v8f acc[4];
#pragma unroll
    for (int r = 0; r < 8; r++) { mrun[r] = -3.0e38f; lrun[r] = 0.0f; }
#pragma unroll
    for (int t = 0; t < 4; t++) acc[t] = (v8f){};

    // Prologue: wave 0 kicks DMA of the first K/V tiles into buffer 0.
    if (threadIdx.x < 32) {
        tdm_load_2d(smem,        Kbase, DKK, SS, DKK, 64, DKK);  // K rows 0..63
        tdm_load_2d(smem + 8192, Vbase, SS, DKK, 64, DKK, SS);   // V cols 0..63
    }

    int buf = 0;
    for (int j0 = 0; j0 < SS; j0 += 64) {
        if (threadIdx.x < 32) __builtin_amdgcn_s_wait_tensorcnt(0);
        __syncthreads();     // publishes: buf ready + everyone done reading buf^1
        if (threadIdx.x < 32 && (j0 + 64) < SS) {
            tdm_load_2d(smem + (buf ^ 1) * 4096, Kbase + (size_t)(j0 + 64) * DKK,
                        DKK, SS, DKK, 64, DKK);
            tdm_load_2d(smem + 8192 + (buf ^ 1) * 4096, Vbase + (j0 + 64),
                        SS, DKK, 64, DKK, SS);
        }
        const __bf16* kb = smem + buf * 4096;
        const __bf16* vb = smem + 8192 + buf * 4096;

        // --- scores: four 16x16 tiles from LDS K fragments ---
        v8f c[4];
        int mk[4];
#pragma unroll
        for (int g = 0; g < 4; g++) {
            const __bf16* krow = kb + (size_t)(16 * g + N) * DKK;
            Frag b0 = mk_frag_bf16(krow, half);
            Frag b1 = mk_frag_bf16(krow + 32, half);
            c[g] = (v8f){};
            c[g] = __builtin_amdgcn_wmma_f32_16x16x32_bf16(false, aq0.v, false, b0.v,
                                                           (short)0, c[g], false, false);
            c[g] = __builtin_amdgcn_wmma_f32_16x16x32_bf16(false, aq1.v, false, b1.v,
                                                           (short)0, c[g], false, false);
            mk[g] = mrow[j0 + 16 * g + N];
        }
        // --- scale + mask + one online-softmax update over 64 columns ---
#pragma unroll
        for (int r = 0; r < 8; r++) {
            float s[4];
#pragma unroll
            for (int g = 0; g < 4; g++)
                s[g] = mk[g] ? c[g][r] * 0.125f : -1e9f;   // 1/sqrt(64)
            float rowmax = redmax16(fmaxf(fmaxf(s[0], s[1]), fmaxf(s[2], s[3])));
            float mnew = fmaxf(mrun[r], rowmax);
            float p[4];
#pragma unroll
            for (int g = 0; g < 4; g++) p[g] = __expf(s[g] - mnew);
            float corr = __expf(mrun[r] - mnew);
            lrun[r] = lrun[r] * corr + redsum16((p[0] + p[1]) + (p[2] + p[3]));
            mrun[r] = mnew;
#pragma unroll
            for (int t = 0; t < 4; t++) acc[t][r] *= corr;
            int row = r + 8 * half;
#pragma unroll
            for (int g = 0; g < 4; g++)
                pl[row * 64 + 16 * g + N] = (__bf16)p[g];  // stage P (C layout -> LDS)
        }
        // --- re-fragment P as two 16x32 A chunks (same-wave LDS, in-order) ---
        Frag ap0, ap1;
        ap0.h[0] = *(const v8bf*)(pl + N * 64 + half * 8);
        ap0.h[1] = *(const v8bf*)(pl + N * 64 + 16 + half * 8);
        ap1.h[0] = *(const v8bf*)(pl + N * 64 + 32 + half * 8);
        ap1.h[1] = *(const v8bf*)(pl + N * 64 + 48 + half * 8);
        // --- O += P @ V  (V fragments from LDS, rows contiguous along s) ---
#pragma unroll
        for (int t = 0; t < 4; t++) {
            const __bf16* vr = vb + (size_t)(t * 16 + N) * DKK;
            Frag bv0 = mk_frag_bf16(vr, half);
            Frag bv1 = mk_frag_bf16(vr + 32, half);
            acc[t] = __builtin_amdgcn_wmma_f32_16x16x32_bf16(false, ap0.v, false, bv0.v,
                                                             (short)0, acc[t], false, false);
            acc[t] = __builtin_amdgcn_wmma_f32_16x16x32_bf16(false, ap1.v, false, bv1.v,
                                                             (short)0, acc[t], false, false);
        }
        buf ^= 1;
    }

    // --- normalize and scatter to mh[b, s, h*64 + d] (bf16 for final GEMM) ---
#pragma unroll
    for (int r = 0; r < 8; r++) {
        float inv = 1.0f / lrun[r];
        int srow = i0 + r + 8 * half;
        size_t orow = ((size_t)b * SS + srow) * DD + h * DKK;
#pragma unroll
        for (int t = 0; t < 4; t++)
            mh[orow + t * 16 + N] = (__bf16)(acc[t][r] * inv);
    }
}

extern "C" void kernel_launch(void* const* d_in, const int* in_sizes, int n_in,
                              void* d_out, int out_size, void* d_ws, size_t ws_size,
                              hipStream_t stream) {
    const float* q  = (const float*)d_in[0];
    const float* k  = (const float*)d_in[1];
    const float* v  = (const float*)d_in[2];
    const int*   mk = (const int*)d_in[3];
    const float* Wq = (const float*)d_in[4];
    const float* Wk = (const float*)d_in[5];
    const float* Wv = (const float*)d_in[6];
    const float* Wo = (const float*)d_in[7];
    float* out = (float*)d_out;

    const size_t NE = (size_t)BS * DD;             // 6,291,456 elements
    const size_t NW = (size_t)DD * DD;             //   589,824 elements
    char* w = (char*)d_ws;                         // needs ~118 MB workspace
    __bf16* qb  = (__bf16*)w;                      // bf16 copies of inputs
    __bf16* kb  = qb + NE;
    __bf16* vb  = kb + NE;
    __bf16* Wqb = vb + NE;
    __bf16* Wkb = Wqb + NW;
    __bf16* Wvb = Wkb + NW;
    __bf16* Wob = Wvb + NW;
    __bf16* Qp  = Wob + NW;                        // bf16 q-projection  [b,s,d]
    __bf16* Kp  = Qp + NE;                         // bf16 k-projection  [b,s,d]
    __bf16* Qr  = Kp + NE;                         // rope(Kp)  [b,h,s,dk]
    __bf16* Kr  = Qr + NE;                         // rope(Qp)  [b,h,s,dk]
    __bf16* Vt  = Kr + NE;                         // v-proj    [b,h,dk,s]
    __bf16* mh  = Vt + NE;                         // attention output [b,s,d]

    // 0) one-time fp32 -> bf16 conversions
    const int ne8 = (int)(NE / 8), nw8 = (int)(NW / 8);
    cvt_bf16<<<(ne8 + 255) / 256, 256, 0, stream>>>(q, qb, ne8);
    cvt_bf16<<<(ne8 + 255) / 256, 256, 0, stream>>>(k, kb, ne8);
    cvt_bf16<<<(ne8 + 255) / 256, 256, 0, stream>>>(v, vb, ne8);
    cvt_bf16<<<(nw8 + 255) / 256, 256, 0, stream>>>(Wq, Wqb, nw8);
    cvt_bf16<<<(nw8 + 255) / 256, 256, 0, stream>>>(Wk, Wkb, nw8);
    cvt_bf16<<<(nw8 + 255) / 256, 256, 0, stream>>>(Wv, Wvb, nw8);
    cvt_bf16<<<(nw8 + 255) / 256, 256, 0, stream>>>(Wo, Wob, nw8);

    dim3 blk(128);
    const int gemm_blocks = (BS / 16) * (DD / 64) / 4;   // 1536 blocks, 1 wave/16x64 tile
    gemm_tile<0><<<gemm_blocks, blk, 0, stream>>>(qb, Wqb, Qp);
    gemm_tile<0><<<gemm_blocks, blk, 0, stream>>>(kb, Wkb, Kp);
    gemm_tile<2><<<gemm_blocks, blk, 0, stream>>>(vb, Wvb, Vt);

    const int rope_threads = BB * HH * SS * 32;          // 3,145,728
    rope_swap<<<rope_threads / 256, 256, 0, stream>>>(Qp, Kp, Qr, Kr);

    const int attn_blocks = BB * HH * (SS / 16) / 4;     // 1536 blocks
    attn_kernel<<<attn_blocks, blk, 0, stream>>>(Qr, Kr, Vt, mk, mh);

    gemm_tile<1><<<gemm_blocks, blk, 0, stream>>>(mh, Wob, out);
}
